// AdaptiveWeightedFConLoss_46342697123848
// MI455X (gfx1250) — matrix operand order repaired and checked by reading, not compiled, not run on hardware
//
#include <hip/hip_runtime.h>
#include <hip/hip_bf16.h>

#define NN 6144
#define DD 128
#define JSPLIT 4

typedef __attribute__((ext_vector_type(16))) __bf16 v16bf;
typedef __attribute__((ext_vector_type(8)))  float  v8f;

union Frag {
    v16bf bf;
    unsigned int u[8];
    uint4 q[2];
};

__device__ __forceinline__ unsigned short f2bf(float x) {
    unsigned int u = __float_as_uint(x);
    unsigned int r = (u + 0x7FFFu + ((u >> 16) & 1u)) >> 16;
    return (unsigned short)r;
}

// Pack two exact-0/1 floats into one dword of two bf16 (bf16 == high 16 bits, no rounding
// needed for 0.0/1.0). Single v_perm_b32.
__device__ __forceinline__ unsigned int pack01(float lo, float hi) {
    return __builtin_amdgcn_perm(__float_as_uint(hi), __float_as_uint(lo), 0x07060302u);
}

// ---------------- K1: L2-normalize + bf16 conversion ----------------
// grid (N, 3), block 128. which: 0=embF->Fn(row major), 1=embM->MnT(T)+eM, 2=embP->PnT(T)+eP
__global__ __launch_bounds__(128) void k1_norm(
    const float* __restrict__ embF, const float* __restrict__ embM,
    const float* __restrict__ embP,
    unsigned short* __restrict__ Fn, unsigned short* __restrict__ MnT,
    unsigned short* __restrict__ PnT, unsigned short* __restrict__ eM,
    unsigned short* __restrict__ eP)
{
    const int row = blockIdx.x;
    const int t = threadIdx.x;
    const int which = blockIdx.y;
    const float* src = (which == 0) ? embF : (which == 1) ? embM : embP;
    float val = src[(size_t)row * DD + t];

    __shared__ float red[128];
    red[t] = val * val;
    __syncthreads();
    #pragma unroll
    for (int s = 64; s > 0; s >>= 1) {
        if (t < s) red[t] += red[t + s];
        __syncthreads();
    }
    float norm = sqrtf(red[0]);
    float nv = val / fmaxf(norm, 1e-12f);

    if (which == 0) {
        Fn[(size_t)row * DD + t] = f2bf(nv);
    } else if (which == 1) {
        MnT[(size_t)row * DD + t] = f2bf(nv);
        eM[(size_t)t * NN + row]  = f2bf(val);
    } else {
        PnT[(size_t)row * DD + t] = f2bf(nv);
        eP[(size_t)t * NN + row]  = f2bf(val);
    }
}

// ---------------- K2: fused WMMA kernel ----------------
// grid (N/64, JSPLIT, 2), block 128 (4 waves). Each wave owns a 16-row i-tile and
// one j-split; reads adjacency exactly once from HBM (wide b128, A-operand layout),
// produces per-split partials:
//   repr = adj @ emb, pos = sum(exp(sim/tau)*adj), tot = sum(exp(sim/tau)), deg = rowsum(adj)
__global__ __launch_bounds__(128) void k2_fused(
    const float* __restrict__ adjFM, const float* __restrict__ adjFP,
    const unsigned short* __restrict__ Fn,
    const unsigned short* __restrict__ MnT, const unsigned short* __restrict__ PnT,
    const unsigned short* __restrict__ eM, const unsigned short* __restrict__ eP,
    float* __restrict__ reprP,  // [2][JSPLIT][N*D]
    float* __restrict__ posP,   // [2][JSPLIT][N]
    float* __restrict__ totP,   // [2][JSPLIT][N]
    float* __restrict__ degP)   // [2][JSPLIT][N]
{
    const int lane = threadIdx.x & 31;
    const int w    = threadIdx.x >> 5;
    const int m    = lane & 15;      // row (A) / col (C) index within tile
    const int h    = lane >> 4;      // half-wave select
    const int i0   = blockIdx.x * 64 + w * 16;
    const int split = blockIdx.y;
    const int jbeg = split * (NN / JSPLIT);
    const int jend = jbeg + (NN / JSPLIT);
    const int z    = blockIdx.z;     // 0 = FM, 1 = FP

    const float* __restrict__ adj          = z ? adjFP : adjFM;
    const unsigned short* __restrict__ T   = z ? PnT : MnT;
    const unsigned short* __restrict__ EMB = z ? eP : eM;

    __shared__ unsigned short tile[4][2][16][16];  // per-wave adjacency mask restage (bf16)

    // Preload Fn A-fragments for the whole j loop: 16 rows x 128 K in 4 chunks of 32.
    Frag afn[4];
    #pragma unroll
    for (int kc = 0; kc < 4; ++kc) {
        const unsigned short* p = Fn + (size_t)(i0 + m) * DD + kc * 32 + 8 * h;
        afn[kc].q[0] = *(const uint4*)(p);
        afn[kc].q[1] = *(const uint4*)(p + 16);
    }

    v8f crepr[8];
    const v8f vzero = {0.f, 0.f, 0.f, 0.f, 0.f, 0.f, 0.f, 0.f};
    #pragma unroll
    for (int cb = 0; cb < 8; ++cb) crepr[cb] = vzero;

    float posAcc[8], totAcc[8];
    #pragma unroll
    for (int v = 0; v < 8; ++v) { posAcc[v] = 0.f; totAcc[v] = 0.f; }
    float degA = 0.f;   // degree accumulates directly in A-layout (row m per lane)

    for (int jc = jbeg; jc < jend; jc += 32) {
        // Prefetch next chunk of the adjacency stream (the only HBM-resident data):
        // one byte per lane at 64B granularity covers rows i0..i0+15, cols jc+32..jc+63.
        if (jc + 32 < jend) {
            __builtin_prefetch(adj + (size_t)(i0 + m) * NN + (jc + 32) + 16 * h, 0, 1);
        }

        // Adjacency chunk (16 rows x 32 cols) loaded in A-operand layout with 4x b128:
        // lane (m,h): tile0 cols jc+8h..jc+8h+7, tile1 cols jc+16+8h..jc+16+8h+7.
        const float* ap = adj + (size_t)(i0 + m) * NN + jc + 8 * h;
        float4 a0 = *(const float4*)(ap);
        float4 a1 = *(const float4*)(ap + 4);
        float4 a2 = *(const float4*)(ap + 16);
        float4 a3 = *(const float4*)(ap + 20);

        degA += (a0.x + a0.y + a0.z + a0.w) + (a1.x + a1.y + a1.z + a1.w)
              + (a2.x + a2.y + a2.z + a2.w) + (a3.x + a3.y + a3.z + a3.w);

        Frag aadj;                       // bf16 A-fragment, exact for 0/1 values
        aadj.u[0] = pack01(a0.x, a0.y);
        aadj.u[1] = pack01(a0.z, a0.w);
        aadj.u[2] = pack01(a1.x, a1.y);
        aadj.u[3] = pack01(a1.z, a1.w);
        aadj.u[4] = pack01(a2.x, a2.y);
        aadj.u[5] = pack01(a2.z, a2.w);
        aadj.u[6] = pack01(a3.x, a3.y);
        aadj.u[7] = pack01(a3.z, a3.w);

        // Restage mask to LDS (wide stores); read back per-element during exp masking.
        *(uint4*)&tile[w][0][m][8 * h] = aadj.q[0];
        *(uint4*)&tile[w][1][m][8 * h] = aadj.q[1];

        // repr += adj(16x32) @ emb(32x128): A straight from registers, 8 column-block WMMAs.
        const unsigned short* ep = EMB + (size_t)m * NN + jc + 8 * h;
        #pragma unroll
        for (int cb = 0; cb < 8; ++cb) {
            Frag b;
            b.q[0] = *(const uint4*)(ep + (size_t)cb * 16 * NN);
            b.q[1] = *(const uint4*)(ep + (size_t)cb * 16 * NN + 16);
            crepr[cb] = __builtin_amdgcn_wmma_f32_16x16x32_bf16(
                            false, aadj.bf, false, b.bf, (short)0, crepr[cb], false, false);
        }

        // sim tiles: Fn(16x128) @ Mn^T(128x16), K=128 in 4 bf16 WMMAs each, f32 accum.
        #pragma unroll
        for (int t = 0; t < 2; ++t) {
            const int j0 = jc + 16 * t;
            v8f c = vzero;
            #pragma unroll
            for (int kc = 0; kc < 4; ++kc) {
                Frag b;
                const unsigned short* bp = T + (size_t)(j0 + m) * DD + kc * 32 + 8 * h;
                b.q[0] = *(const uint4*)(bp);
                b.q[1] = *(const uint4*)(bp + 16);
                c = __builtin_amdgcn_wmma_f32_16x16x32_bf16(
                        false, afn[kc].bf, false, b.bf, (short)0, c, false, false);
            }
            // exp + masked accumulation; mask from LDS restage (C-layout indexing)
            #pragma unroll
            for (int v = 0; v < 8; ++v) {
                float e = __expf(c[v] * 10.0f);   // 1/TAU = 10
                totAcc[v] += e;
                posAcc[v] += (tile[w][t][v + 8 * h][m] != 0) ? e : 0.0f;
            }
        }
    }

    const size_t vecBase = ((size_t)z * JSPLIT + split) * NN;

    // Degree: lane m and lane m+16 jointly cover all 32 columns of every chunk.
    degA += __shfl_xor(degA, 16);
    if (h == 0) degP[vecBase + i0 + m] = degA;

    // pos/tot row sums across the 16 column-lanes of each half-wave.
    #pragma unroll
    for (int v = 0; v < 8; ++v) {
        #pragma unroll
        for (int s = 8; s >= 1; s >>= 1) {
            posAcc[v] += __shfl_xor(posAcc[v], s);
            totAcc[v] += __shfl_xor(totAcc[v], s);
        }
    }
    if (m == 0) {  // lanes 0 and 16: rows 0-7 and 8-15
        #pragma unroll
        for (int v = 0; v < 8; ++v) {
            const int row = i0 + v + 8 * h;
            posP[vecBase + row] = posAcc[v];
            totP[vecBase + row] = totAcc[v];
        }
    }
    // repr partial write (each element owned by exactly one lane -> no atomics)
    float* rp = reprP + ((size_t)z * JSPLIT + split) * ((size_t)NN * DD);
    #pragma unroll
    for (int cb = 0; cb < 8; ++cb) {
        #pragma unroll
        for (int v = 0; v < 8; ++v)
            rp[(size_t)(i0 + v + 8 * h) * DD + cb * 16 + m] = crepr[cb][v];
    }
}

// ---------------- K3: MLP + softmax -> weights ----------------
// grid N, block 128
__global__ __launch_bounds__(128) void k3_mlp(
    const float* __restrict__ reprP, const float* __restrict__ degP,
    const float* __restrict__ W1, const float* __restrict__ b1,
    const float* __restrict__ W2, const float* __restrict__ b2,
    float* __restrict__ out)
{
    const int r = blockIdx.x;
    const int t = threadIdx.x;
    __shared__ float feat[256];

    float dM = 0.f, dP = 0.f, fm = 0.f, fp = 0.f;
    #pragma unroll
    for (int s = 0; s < JSPLIT; ++s) {
        dM += degP[(size_t)(0 * JSPLIT + s) * NN + r];
        dP += degP[(size_t)(1 * JSPLIT + s) * NN + r];
        fm += reprP[((size_t)(0 * JSPLIT + s)) * ((size_t)NN * DD) + (size_t)r * DD + t];
        fp += reprP[((size_t)(1 * JSPLIT + s)) * ((size_t)NN * DD) + (size_t)r * DD + t];
    }
    feat[t]       = fm / fmaxf(dM, 1.0f);
    feat[128 + t] = fp / fmaxf(dP, 1.0f);
    __syncthreads();

    float hacc = b1[t];
    #pragma unroll 8
    for (int k = 0; k < 256; ++k) hacc += feat[k] * W1[(size_t)k * DD + t];
    hacc = fmaxf(hacc, 0.0f);

    __shared__ float r0[128], r1[128];
    r0[t] = hacc * W2[t * 2 + 0];
    r1[t] = hacc * W2[t * 2 + 1];
    __syncthreads();
    #pragma unroll
    for (int s = 64; s > 0; s >>= 1) {
        if (t < s) { r0[t] += r0[t + s]; r1[t] += r1[t + s]; }
        __syncthreads();
    }
    if (t == 0) {
        float l0 = r0[0] + b2[0], l1 = r1[0] + b2[1];
        float mx = fmaxf(l0, l1);
        float e0 = __expf(l0 - mx), e1 = __expf(l1 - mx);
        float sum = e0 + e1;
        out[1 + 2 * r]     = e0 / sum;
        out[1 + 2 * r + 1] = e1 / sum;
    }
}

// ---------------- K4: loss reduction (single block, deterministic) ----------------
__global__ __launch_bounds__(256) void k4_loss(
    const float* __restrict__ posP, const float* __restrict__ totP,
    const float* __restrict__ degP, float* __restrict__ out)
{
    __shared__ float red[256];
    float acc = 0.f;
    for (int r = threadIdx.x; r < NN; r += 256) {
        float pM = 0.f, pP = 0.f, tM = 0.f, tP = 0.f, dM = 0.f, dP = 0.f;
        #pragma unroll
        for (int s = 0; s < JSPLIT; ++s) {
            pM += posP[(size_t)(0 * JSPLIT + s) * NN + r];
            pP += posP[(size_t)(1 * JSPLIT + s) * NN + r];
            tM += totP[(size_t)(0 * JSPLIT + s) * NN + r];
            tP += totP[(size_t)(1 * JSPLIT + s) * NN + r];
            dM += degP[(size_t)(0 * JSPLIT + s) * NN + r];
            dP += degP[(size_t)(1 * JSPLIT + s) * NN + r];
        }
        float w0 = out[1 + 2 * r], w1 = out[2 + 2 * r];
        float wp = w0 * pM + w1 * pP;
        float wn = w0 * (tM - pM) + w1 * (tP - pP);
        float nei = fmaxf(dM + dP, 1.0f);
        float ratio = fmaxf(wp / (wp + wn) / nei, 1e-10f);
        acc += -logf(ratio);
    }
    red[threadIdx.x] = acc;
    __syncthreads();
    #pragma unroll
    for (int s = 128; s > 0; s >>= 1) {
        if (threadIdx.x < s) red[threadIdx.x] += red[threadIdx.x + s];
        __syncthreads();
    }
    if (threadIdx.x == 0) out[0] = red[0] / (float)NN;
}

// ---------------- launch ----------------
extern "C" void kernel_launch(void* const* d_in, const int* in_sizes, int n_in,
                              void* d_out, int out_size, void* d_ws, size_t ws_size,
                              hipStream_t stream) {
    const float* embF  = (const float*)d_in[0];
    const float* embM  = (const float*)d_in[1];
    const float* embP  = (const float*)d_in[2];
    const float* adjFM = (const float*)d_in[3];
    const float* adjFP = (const float*)d_in[4];
    const float* W1    = (const float*)d_in[5];
    const float* b1    = (const float*)d_in[6];
    const float* W2    = (const float*)d_in[7];
    const float* b2    = (const float*)d_in[8];
    float* out = (float*)d_out;

    char* ws = (char*)d_ws;
    size_t o = 0;
    const size_t bf = (size_t)NN * DD * sizeof(unsigned short);   // 1.5 MB each
    unsigned short* Fn  = (unsigned short*)(ws + o); o += bf;
    unsigned short* MnT = (unsigned short*)(ws + o); o += bf;
    unsigned short* PnT = (unsigned short*)(ws + o); o += bf;
    unsigned short* eM  = (unsigned short*)(ws + o); o += bf;
    unsigned short* eP  = (unsigned short*)(ws + o); o += bf;
    float* reprP = (float*)(ws + o); o += (size_t)2 * JSPLIT * NN * DD * sizeof(float);
    float* posP  = (float*)(ws + o); o += (size_t)2 * JSPLIT * NN * sizeof(float);
    float* totP  = (float*)(ws + o); o += (size_t)2 * JSPLIT * NN * sizeof(float);
    float* degP  = (float*)(ws + o); o += (size_t)2 * JSPLIT * NN * sizeof(float);

    k1_norm<<<dim3(NN, 3), 128, 0, stream>>>(embF, embM, embP, Fn, MnT, PnT, eM, eP);
    k2_fused<<<dim3(NN / 64, JSPLIT, 2), 128, 0, stream>>>(
        adjFM, adjFP, Fn, MnT, PnT, eM, eP, reprP, posP, totP, degP);
    k3_mlp<<<dim3(NN), 128, 0, stream>>>(reprP, degP, W1, b1, W2, b2, out);
    k4_loss<<<1, 256, 0, stream>>>(posP, totP, degP, out);
}
